// MyGRU_52621939310799
// MI455X (gfx1250) — compile-verified, matrix operand order checked by reading, not run on hardware
//
#include <hip/hip_runtime.h>
#include <stdint.h>

// ---------------------------------------------------------------------------
// GRU on gfx1250 (wave32):
//   - one-time f32->bf16 conversion of X and all weights (L2-resident reuse)
//   - bf16 WMMA GEMM core: block tile 64x128, wave tile 32x32 (4 WMMA/K-step)
//   - LDS staging via GLOBAL_LOAD_ASYNC_TO_LDS_B128 (ASYNCcnt), double-buffered
//     with STATIC ping-pong addressing (K-loop unrolled by 2), 1 barrier/K-step
//   - recurrence: 512 steps x 2 kernels (z,r -> h_cand/update), h ping-pong
// ---------------------------------------------------------------------------

typedef __attribute__((ext_vector_type(16))) __bf16 v16bf;
typedef __attribute__((ext_vector_type(8)))  __bf16 v8bf;
typedef __attribute__((ext_vector_type(8)))  float  v8f;
typedef __attribute__((ext_vector_type(4)))  float  v4f;

#define LDSS   40               // LDS row stride (bf16 elems): 80B rows, 16B-aligned chunks
#define A_TILE (64  * LDSS)     // one A buffer (64 rows)
#define B_TILE (128 * LDSS)     // one B buffer (128 n-rows)

union FragBF16 { v16bf v; uint32_t u[8]; };

__device__ __forceinline__ uint32_t lds_off(const void* p) {
    // generic LDS pointer: low 32 bits are the LDS byte address
    return (uint32_t)(uintptr_t)p;
}

__device__ __forceinline__ void async_b128(uint32_t ldsaddr, const void* gaddr) {
    asm volatile("global_load_async_to_lds_b128 %0, %1, off"
                 :: "v"(ldsaddr), "v"(gaddr) : "memory");
}

__device__ __forceinline__ void wait_async0() {
    asm volatile("s_wait_asynccnt 0x0" ::: "memory");
}

// Stage A (64x32) and B (128x32) bf16 tiles for K-slice kk: 3 async b128 per wave.
__device__ __forceinline__ void issue_tiles(const __bf16* __restrict__ A,
                                            const __bf16* __restrict__ W,
                                            int K, int rowBase, int colBase, int kk,
                                            __bf16* AsB, __bf16* BsB, int tid)
{
    const int ar = tid >> 2, ac = (tid & 3) << 3;       // 64 rows x 4 chunks
    async_b128(lds_off(AsB + ar * LDSS + ac),
               A + (size_t)(rowBase + ar) * K + kk + ac);
    #pragma unroll
    for (int e = 0; e < 2; ++e) {
        const int cid = tid + e * 256;                   // 128 rows x 4 chunks
        const int br = cid >> 2, bc = (cid & 3) << 3;
        async_b128(lds_off(BsB + br * LDSS + bc),
                   W + (size_t)(colBase + br) * K + kk + bc);
    }
}

// one K-step of the 32x32 wave tile: 2 A-frags x 2 B-frags -> 4 WMMAs
__device__ __forceinline__ void compute_32x32(const __bf16* AsC, const __bf16* BsC,
                                              int wm, int wn, int half, int lrow,
                                              v8f acc[2][2])
{
    FragBF16 af[2], bfm[2];
    #pragma unroll
    for (int i = 0; i < 2; ++i) {
        const uint32_t* ap = (const uint32_t*)(AsC + (wm * 32 + i * 16 + lrow) * LDSS);
        const uint32_t* bp = (const uint32_t*)(BsC + (wn * 32 + i * 16 + lrow) * LDSS);
        #pragma unroll
        for (int j = 0; j < 8; ++j) {
            const int kb = (j < 4) ? (half * 8 + 2 * j) : (16 + half * 8 + 2 * (j - 4));
            af[i].u[j]  = ap[kb >> 1];
            bfm[i].u[j] = bp[kb >> 1];
        }
    }
    #pragma unroll
    for (int i = 0; i < 2; ++i)
        #pragma unroll
        for (int j = 0; j < 2; ++j)
            acc[i][j] = __builtin_amdgcn_wmma_f32_16x16x32_bf16(
                false, af[i].v, false, bfm[j].v, (short)0, acc[i][j], false, false);
}

// acc[2][2] += A[rowBase..+63, :] @ W[colBase..+127, :]^T  (bf16 in, f32 acc)
// K/32 must be even (K = 512 or 1024 here) -> statically addressed ping-pong.
__device__ __forceinline__ void wmma_gemm_64x128(const __bf16* __restrict__ A,
                                                 const __bf16* __restrict__ W,
                                                 int K, int rowBase, int colBase,
                                                 __bf16* As0, __bf16* Bs0,
                                                 __bf16* As1, __bf16* Bs1,
                                                 v8f acc[2][2])
{
    const int tid  = threadIdx.x;
    const int wave = tid >> 5, lane = tid & 31;
    const int wm   = wave & 1, wn = wave >> 1;          // wave tile origin
    const int half = lane >> 4, lrow = lane & 15;
    const int iters = K >> 5;                           // even

    issue_tiles(A, W, K, rowBase, colBase, 0, As0, Bs0, tid);
    for (int i = 0; i < iters; i += 2) {
        // ---- even K-step: compute buf0, prefetch (i+1) into buf1 ----
        wait_async0();                // buf0 landed
        __syncthreads();              // all stores visible; all prior buf1 reads done
        issue_tiles(A, W, K, rowBase, colBase, (i + 1) << 5, As1, Bs1, tid);
        compute_32x32(As0, Bs0, wm, wn, half, lrow, acc);
        // ---- odd K-step: compute buf1, prefetch (i+2) into buf0 ----
        wait_async0();                // buf1 landed
        __syncthreads();              // all prior buf0 reads done
        if (i + 2 < iters)
            issue_tiles(A, W, K, rowBase, colBase, (i + 2) << 5, As0, Bs0, tid);
        compute_32x32(As1, Bs1, wm, wn, half, lrow, acc);
    }
}

// elementwise f32 -> bf16 (n multiple of 8)
__global__ __launch_bounds__(256) void cvt_bf16_kernel(const float* __restrict__ src,
                                                       __bf16* __restrict__ dst, int n)
{
    const int i = (blockIdx.x * 256 + threadIdx.x) * 8;
    if (i >= n) return;
    v4f a = *(const v4f*)(src + i);
    v4f b = *(const v4f*)(src + i + 4);
    v8bf cv;
    cv[0]=(__bf16)a[0]; cv[1]=(__bf16)a[1]; cv[2]=(__bf16)a[2]; cv[3]=(__bf16)a[3];
    cv[4]=(__bf16)b[0]; cv[5]=(__bf16)b[1]; cv[6]=(__bf16)b[2]; cv[7]=(__bf16)b[3];
    *(v8bf*)(dst + i) = cv;
}

// Y[M,N] = Xbf[M,K] @ Wbf[N,K]^T + bias   (f32 out)
__global__ __launch_bounds__(256) void proj_kernel(const __bf16* __restrict__ X,
                                                   const __bf16* __restrict__ W,
                                                   const float* __restrict__ bias,
                                                   float* __restrict__ Y, int N, int K)
{
    __shared__ __bf16 As[2 * A_TILE];
    __shared__ __bf16 Bs[2 * B_TILE];
    const int rowBase = blockIdx.y * 64;
    const int colBase = blockIdx.x * 128;
    v8f acc[2][2] = {};
    wmma_gemm_64x128(X, W, K, rowBase, colBase,
                     As, Bs, As + A_TILE, Bs + B_TILE, acc);

    const int lane = threadIdx.x & 31, wave = threadIdx.x >> 5;
    const int wm = wave & 1, wn = wave >> 1;
    const int half = lane >> 4, lcol = lane & 15;
    #pragma unroll
    for (int j = 0; j < 2; ++j) {
        const int col = colBase + wn * 32 + j * 16 + lcol;
        const float bv = bias[col];
        #pragma unroll
        for (int i = 0; i < 2; ++i)
            #pragma unroll
            for (int e = 0; e < 8; ++e) {
                const int row = rowBase + wm * 32 + i * 16 + half * 8 + e;
                Y[(size_t)row * N + col] = acc[i][j][e] + bv;
            }
    }
}

// gate kernel: blockIdx.y==0 -> z = sigmoid(.) (f32)
//              blockIdx.y==1 -> rh = bf16(sigmoid(.) * hprev)
__global__ __launch_bounds__(256) void gru_zr_kernel(
    const __bf16* __restrict__ hbf,
    const __bf16* __restrict__ Wz, const float* __restrict__ bz,
    const __bf16* __restrict__ Wr, const float* __restrict__ brr,
    const float* __restrict__ xz_t, const float* __restrict__ xr_t,
    const float* __restrict__ hf,
    float* __restrict__ zout, __bf16* __restrict__ rhbf, int N, int K)
{
    __shared__ __bf16 As[2 * A_TILE];
    __shared__ __bf16 Bs[2 * B_TILE];
    const bool isZ = (blockIdx.y == 0);
    const __bf16* W    = isZ ? Wz : Wr;
    const float*  bias = isZ ? bz : brr;
    const float*  x    = isZ ? xz_t : xr_t;
    const int colBase = blockIdx.x * 128;
    v8f acc[2][2] = {};
    wmma_gemm_64x128(hbf, W, K, 0, colBase,
                     As, Bs, As + A_TILE, Bs + B_TILE, acc);

    const int lane = threadIdx.x & 31, wave = threadIdx.x >> 5;
    const int wm = wave & 1, wn = wave >> 1;
    const int half = lane >> 4, lcol = lane & 15;
    #pragma unroll
    for (int j = 0; j < 2; ++j) {
        const int col = colBase + wn * 32 + j * 16 + lcol;
        const float bv = bias[col];
        #pragma unroll
        for (int i = 0; i < 2; ++i)
            #pragma unroll
            for (int e = 0; e < 8; ++e) {
                const int row = wm * 32 + i * 16 + half * 8 + e;   // batch index
                const size_t idx = (size_t)row * N + col;
                const float g = 1.0f / (1.0f + __expf(-(acc[i][j][e] + bv + x[idx])));
                if (isZ) zout[idx] = g;
                else     rhbf[idx] = (__bf16)(g * hf[idx]);
            }
    }
}

// h_cand = tanh(rh @ Whh^T + b + xh_t); h_new = (1-z)*h + z*h_cand; mask by length
__global__ __launch_bounds__(256) void gru_h_kernel(
    const __bf16* __restrict__ rhbf,
    const __bf16* __restrict__ Wh, const float* __restrict__ bh,
    const float* __restrict__ xh_t, const float* __restrict__ zgate,
    const float* __restrict__ hf, const long long* __restrict__ lengths, int t,
    float* __restrict__ hfnext, __bf16* __restrict__ hbfnext,
    float* __restrict__ hstack_t, float* __restrict__ hlast, int N, int K)
{
    __shared__ __bf16 As[2 * A_TILE];
    __shared__ __bf16 Bs[2 * B_TILE];
    const int colBase = blockIdx.x * 128;
    v8f acc[2][2] = {};
    wmma_gemm_64x128(rhbf, Wh, K, 0, colBase,
                     As, Bs, As + A_TILE, Bs + B_TILE, acc);

    const int lane = threadIdx.x & 31, wave = threadIdx.x >> 5;
    const int wm = wave & 1, wn = wave >> 1;
    const int half = lane >> 4, lcol = lane & 15;
    #pragma unroll
    for (int j = 0; j < 2; ++j) {
        const int col = colBase + wn * 32 + j * 16 + lcol;
        const float bv = bh[col];
        #pragma unroll
        for (int i = 0; i < 2; ++i)
            #pragma unroll
            for (int e = 0; e < 8; ++e) {
                const int row = wm * 32 + i * 16 + half * 8 + e;   // batch index
                const size_t idx = (size_t)row * N + col;
                const float hc = tanhf(acc[i][j][e] + bv + xh_t[idx]);
                const float zv = zgate[idx];
                const float hp = hf[idx];
                float hn = (1.0f - zv) * hp + zv * hc;
                hn = ((long long)t < lengths[row]) ? hn : hp;
                hfnext[idx]   = hn;
                hbfnext[idx]  = (__bf16)hn;
                hstack_t[idx] = hn;
                hlast[idx]    = hn;    // final step's write == h_last
            }
    }
}

extern "C" void kernel_launch(void* const* d_in, const int* in_sizes, int n_in,
                              void* d_out, int out_size, void* d_ws, size_t ws_size,
                              hipStream_t stream) {
    (void)in_sizes; (void)n_in; (void)out_size; (void)ws_size;
    const int T = 512, B = 64, NINP = 512, NHID = 1024;

    const float* input = (const float*)d_in[0];
    const float* h0    = (const float*)d_in[1];
    const float* Wiz = (const float*)d_in[2];  const float* biz = (const float*)d_in[3];
    const float* Wir = (const float*)d_in[4];  const float* bir = (const float*)d_in[5];
    const float* Wih = (const float*)d_in[6];  const float* bih = (const float*)d_in[7];
    const float* Whz = (const float*)d_in[8];  const float* bhz = (const float*)d_in[9];
    const float* Whr = (const float*)d_in[10]; const float* bhr = (const float*)d_in[11];
    const float* Whh = (const float*)d_in[12]; const float* bhh = (const float*)d_in[13];
    const long long* lengths = (const long long*)d_in[14];

    const size_t TBH = (size_t)T * B * NHID;
    const size_t BH  = (size_t)B * NHID;
    const size_t XN  = (size_t)T * B * NINP;
    const size_t WIN = (size_t)NHID * NINP;
    const size_t WHN = (size_t)NHID * NHID;

    // f32 workspace region
    float* ws  = (float*)d_ws;
    float* xz  = ws;
    float* xr  = xz + TBH;
    float* xh  = xr + TBH;
    float* zb  = xh + TBH;
    float* hf0 = zb + BH;
    float* hf1 = hf0 + BH;
    // bf16 workspace region (16B aligned: preceding f32 count is a multiple of 4)
    __bf16* wsb   = (__bf16*)(hf1 + BH);
    __bf16* Xbf   = wsb;
    __bf16* Wizb  = Xbf  + XN;
    __bf16* Wirb  = Wizb + WIN;
    __bf16* Wihb  = Wirb + WIN;
    __bf16* Whzb  = Wihb + WIN;
    __bf16* Whrb  = Whzb + WHN;
    __bf16* Whhb  = Whrb + WHN;
    __bf16* hbf0  = Whhb + WHN;
    __bf16* hbf1  = hbf0 + BH;
    __bf16* rhbf  = hbf1 + BH;

    float* hstack = (float*)d_out;       // [T, B, NHID]
    float* hlast  = hstack + TBH;        // [1, B, NHID]

    const dim3 blk(256, 1, 1);

    // one-time conversions (bf16 operands; weights stay L2-resident across steps)
    cvt_bf16_kernel<<<dim3(XN  / 2048), blk, 0, stream>>>(input, Xbf,  (int)XN);
    cvt_bf16_kernel<<<dim3(WIN / 2048), blk, 0, stream>>>(Wiz,  Wizb, (int)WIN);
    cvt_bf16_kernel<<<dim3(WIN / 2048), blk, 0, stream>>>(Wir,  Wirb, (int)WIN);
    cvt_bf16_kernel<<<dim3(WIN / 2048), blk, 0, stream>>>(Wih,  Wihb, (int)WIN);
    cvt_bf16_kernel<<<dim3(WHN / 2048), blk, 0, stream>>>(Whz,  Whzb, (int)WHN);
    cvt_bf16_kernel<<<dim3(WHN / 2048), blk, 0, stream>>>(Whr,  Whrb, (int)WHN);
    cvt_bf16_kernel<<<dim3(WHN / 2048), blk, 0, stream>>>(Whh,  Whhb, (int)WHN);
    cvt_bf16_kernel<<<dim3(BH  / 2048), blk, 0, stream>>>(h0,   hbf0, (int)BH);
    hipMemcpyAsync(hf0, h0, BH * sizeof(float), hipMemcpyDeviceToDevice, stream);

    // phase 1: input projections (M = T*B), block tile 64x128
    {
        const dim3 grd(NHID / 128, (T * B) / 64, 1);
        proj_kernel<<<grd, blk, 0, stream>>>(Xbf, Wizb, biz, xz, NHID, NINP);
        proj_kernel<<<grd, blk, 0, stream>>>(Xbf, Wirb, bir, xr, NHID, NINP);
        proj_kernel<<<grd, blk, 0, stream>>>(Xbf, Wihb, bih, xh, NHID, NINP);
    }

    // phase 2: recurrence (ping-pong f32 + bf16 h buffers)
    float*  hfc = hf0;  float*  hfn = hf1;
    __bf16* hbc = hbf0; __bf16* hbn = hbf1;
    for (int t = 0; t < T; ++t) {
        const float* xz_t = xz + (size_t)t * BH;
        const float* xr_t = xr + (size_t)t * BH;
        const float* xh_t = xh + (size_t)t * BH;
        gru_zr_kernel<<<dim3(NHID / 128, 2), blk, 0, stream>>>(
            hbc, Whzb, bhz, Whrb, bhr, xz_t, xr_t, hfc, zb, rhbf, NHID, NHID);
        gru_h_kernel<<<dim3(NHID / 128, 1), blk, 0, stream>>>(
            rhbf, Whhb, bhh, xh_t, zb, hfc, lengths, t,
            hfn, hbn, hstack + (size_t)t * BH, hlast, NHID, NHID);
        float*  tf = hfc; hfc = hfn; hfn = tf;
        __bf16* tb = hbc; hbc = hbn; hbn = tb;
    }
}
